// Qwen3Attention_20529943675569
// MI455X (gfx1250) — matrix operand order repaired and checked by reading, not compiled, not run on hardware
//
#include <hip/hip_runtime.h>

// ---------------------------------------------------------------------------
// Qwen3 attention for MI455X (gfx1250): bf16 WMMA pipeline
//   convert(+weight transpose) -> QKV GEMMs -> RMSNorm+RoPE -> flash attn -> O GEMM
// Matmuls: v_wmma_f32_16x16x32_bf16 (fp32 accumulate).
// Tile staging: global_load_async_to_lds_b128 (ASYNCcnt) where layout permits.
// ---------------------------------------------------------------------------

typedef __bf16 bf16_t;
typedef __attribute__((ext_vector_type(8)))  __bf16 v8bf;
typedef __attribute__((ext_vector_type(16))) __bf16 v16bf;
typedef __attribute__((ext_vector_type(8)))  float  v8f;

static constexpr int Bn   = 2;
static constexpr int Sn   = 2048;
static constexpr int HIDn = 2048;
static constexpr int Hn   = 16;
static constexpr int KVHn = 8;
static constexpr int HDn  = 128;
static constexpr int Mrows = Bn * Sn;              // 4096
static constexpr float SCALING = 0.088388347648318447f; // 1/sqrt(128)
static constexpr float NEGBIG  = -1e9f;

// ---- CDNA5 async global->LDS copy (16B per lane, tracked by ASYNCcnt) ------
__device__ __forceinline__ void async_copy_b128(const void* gptr, void* lptr) {
  // LDS aperture: generic addr[31:0] == LDS byte offset (ISA 10.2)
  unsigned lds = (unsigned)(size_t)lptr;
  asm volatile("global_load_async_to_lds_b128 %0, %1, off"
               :: "v"(lds), "v"(gptr) : "memory");
}
__device__ __forceinline__ void wait_async0() {
  asm volatile("s_wait_asynccnt 0" ::: "memory");
}

// ---- WMMA fragment loaders (per CDNA5 ISA 7.12.2 layouts) -----------------
// A matrix 16x32 bf16: lane 0-15 -> row=lane, k = {0..7} then {16..23}
//                      lane 16-31 -> row=lane-16, k = {8..15} then {24..31}
__device__ __forceinline__ v16bf frag_a(const bf16_t* base, int stride, int lane) {
  int r    = lane & 15;
  int koff = (lane < 16) ? 0 : 8;
  const bf16_t* p = base + r * stride + koff;
  v8bf lo = *(const v8bf*)(p);
  v8bf hi = *(const v8bf*)(p + 16);
  return __builtin_shufflevector(lo, hi, 0,1,2,3,4,5,6,7,8,9,10,11,12,13,14,15);
}

// B matrix 32x16 bf16 (stored as [n][k], contiguous in k):
//   lane 0-15 -> col=lane, k = {0..7} then {16..23} ; lane 16-31 -> col=lane-16, k = {8..15} then {24..31}
__device__ __forceinline__ v16bf frag_b(const bf16_t* base, int stride, int lane) {
  int n    = lane & 15;
  int koff = (lane < 16) ? 0 : 8;
  const bf16_t* p = base + n * stride + koff;
  v8bf lo = *(const v8bf*)(p);
  v8bf hi = *(const v8bf*)(p + 16);
  return __builtin_shufflevector(lo, hi, 0,1,2,3,4,5,6,7,8,9,10,11,12,13,14,15);
}

// ---- fp32 -> bf16 convert --------------------------------------------------
__global__ void __launch_bounds__(256)
f32_to_bf16_kernel(const float* __restrict__ x, bf16_t* __restrict__ y, int n) {
  int i = blockIdx.x * 256 + threadIdx.x;
  if (i < n) y[i] = (bf16_t)x[i];
}

// ---- fp32 [KxN] -> bf16 [NxK] transpose (one-time, LDS-tiled) --------------
__global__ void __launch_bounds__(256)
transpose_to_bf16_kernel(const float* __restrict__ w, bf16_t* __restrict__ wt,
                         int K, int N) {
  __shared__ bf16_t tile[64][64 + 4];
  int kb = blockIdx.x * 64, nb = blockIdx.y * 64;
  for (int c = threadIdx.x; c < 4096; c += 256) {
    int r = c >> 6, n = c & 63;       // r = k-in-tile; coalesced along n
    tile[r][n] = (bf16_t)w[(size_t)(kb + r) * N + nb + n];
  }
  __syncthreads();
  for (int c = threadIdx.x; c < 4096; c += 256) {
    int r = c >> 6, kk = c & 63;      // r = n-in-tile; coalesced along k
    wt[(size_t)(nb + r) * K + kb + kk] = tile[kk][r];
  }
}

// ---- Tiled bf16 GEMM: C[MxN] = A[MxK] * Bt[NxK]^T, fp32 accumulate --------
// Block tile 128x128, K-step 32. 8 waves; wave w owns rows [16w,16w+16) x 128.
// Both tiles staged with async global->LDS b128 copies.
template<bool OUT_BF16>
__global__ void __launch_bounds__(256)
gemm_bf16_tn_kernel(const bf16_t* __restrict__ A, const bf16_t* __restrict__ Bt,
                    void* __restrict__ Cout, int M, int N, int K) {
  __shared__ __attribute__((aligned(16))) bf16_t lds_a[128 * 32];   // [m][k]
  __shared__ __attribute__((aligned(16))) bf16_t lds_b[128 * 32];   // [n][k]
  int tid  = threadIdx.x;
  int wave = tid >> 5, lane = tid & 31;
  int mb = blockIdx.x * 128, nb = blockIdx.y * 128;

  v8f acc[8] = {};
  for (int k0 = 0; k0 < K; k0 += 32) {
    for (int c = tid; c < 512; c += 256) {      // 128 rows x 4 chunks of 8 bf16
      int r = c >> 2, kk = (c & 3) * 8;
      async_copy_b128(&A[(size_t)(mb + r) * K + k0 + kk],  &lds_a[r * 32 + kk]);
      async_copy_b128(&Bt[(size_t)(nb + r) * K + k0 + kk], &lds_b[r * 32 + kk]);
    }
    wait_async0();
    __syncthreads();

    v16bf af = frag_a(&lds_a[(wave * 16) * 32], 32, lane);
#pragma unroll
    for (int nt = 0; nt < 8; nt++) {
      v16bf bf = frag_b(&lds_b[(nt * 16) * 32], 32, lane);
      acc[nt] = __builtin_amdgcn_wmma_f32_16x16x32_bf16(
          false, af, false, bf, (short)0, acc[nt], false, false);
    }
    __syncthreads();
  }

  // C layout: element e -> row = e + 8*(lane>=16), col = lane&15
  int col = lane & 15, rsel = (lane >> 4) * 8;
#pragma unroll
  for (int nt = 0; nt < 8; nt++) {
#pragma unroll
    for (int e = 0; e < 8; e++) {
      size_t m = (size_t)(mb + wave * 16 + rsel + e);
      size_t n = (size_t)(nb + nt * 16 + col);
      if (OUT_BF16) ((bf16_t*)Cout)[m * N + n] = (bf16_t)acc[nt][e];
      else          ((float*)Cout)[m * N + n]  = acc[nt][e];
    }
  }
}

// ---- RMSNorm + RoPE, wave-per-row ------------------------------------------
// in:  (B, S, NH, HD) bf16   out: (B, NH, S, HD) bf16
// Each lane owns d, d+32, d+64, d+96 -> rotate-half partner is in-lane.
__global__ void __launch_bounds__(256)
rmsnorm_rope_kernel(const bf16_t* __restrict__ in, const float* __restrict__ cosb,
                    const float* __restrict__ sinb, const float* __restrict__ w,
                    bf16_t* __restrict__ out, int NH) {
  int wave = threadIdx.x >> 5, lane = threadIdx.x & 31;
  size_t row = (size_t)blockIdx.x * 8 + wave;       // (b*S + s)*NH + h
  int h = (int)(row % NH);
  size_t bs = row / NH;
  int s = (int)(bs % Sn);
  int b = (int)(bs / Sn);

  const bf16_t* src = in + row * HDn;
  float v0 = (float)src[lane],      v1 = (float)src[lane + 32];
  float v2 = (float)src[lane + 64], v3 = (float)src[lane + 96];
  float ss = v0 * v0 + v1 * v1 + v2 * v2 + v3 * v3;
  ss += __shfl_xor(ss, 1, 32);  ss += __shfl_xor(ss, 2, 32);
  ss += __shfl_xor(ss, 4, 32);  ss += __shfl_xor(ss, 8, 32);
  ss += __shfl_xor(ss, 16, 32);
  float inv = rsqrtf(ss * (1.0f / 128.0f) + 1e-6f);

  float n0 = v0 * inv * w[lane],      n1 = v1 * inv * w[lane + 32];
  float n2 = v2 * inv * w[lane + 64], n3 = v3 * inv * w[lane + 96];

  const float* cp = cosb + ((size_t)b * Sn + s) * HDn;
  const float* sp = sinb + ((size_t)b * Sn + s) * HDn;
  float o0 = n0 * cp[lane]      - n2 * sp[lane];        // d < 64: -x[d+64]*sin
  float o1 = n1 * cp[lane + 32] - n3 * sp[lane + 32];
  float o2 = n2 * cp[lane + 64] + n0 * sp[lane + 64];   // d >= 64: +x[d-64]*sin
  float o3 = n3 * cp[lane + 96] + n1 * sp[lane + 96];

  bf16_t* dst = out + (((size_t)b * NH + h) * Sn + s) * HDn;
  dst[lane]      = (bf16_t)o0;  dst[lane + 32] = (bf16_t)o1;
  dst[lane + 64] = (bf16_t)o2;  dst[lane + 96] = (bf16_t)o3;
}

// ---- V relayout: (B,S,KVH,HD) -> (B,KVH,S,HD) ------------------------------
__global__ void __launch_bounds__(256)
v_relayout_kernel(const bf16_t* __restrict__ in, bf16_t* __restrict__ out) {
  size_t total = (size_t)Bn * Sn * KVHn * HDn;
  size_t i = (size_t)blockIdx.x * 256 + threadIdx.x;
  if (i >= total) return;
  int d = (int)(i & (HDn - 1));
  size_t t = i >> 7;
  int hh = (int)(t % KVHn); t /= KVHn;
  int s  = (int)(t % Sn);
  int b  = (int)(t / Sn);
  out[(((size_t)b * KVHn + hh) * Sn + s) * HDn + d] = in[i];
}

// ---- Flash attention --------------------------------------------------------
// grid = (B*H, S/128); 8 waves/block; wave owns 16 query rows.
// Q fragments held in registers; K tile (async) + V^T tile staged in LDS.
__global__ void __launch_bounds__(256)
flash_attn_kernel(const bf16_t* __restrict__ Q, const bf16_t* __restrict__ K,
                  const bf16_t* __restrict__ V, bf16_t* __restrict__ O) {
  __shared__ __attribute__((aligned(16))) bf16_t k_tile[32 * 128];  // [key][d]
  __shared__ __attribute__((aligned(16))) bf16_t vt[128 * 32];      // [d][key]
  __shared__ __attribute__((aligned(16))) bf16_t p_lds[8][16 * 32]; // per-wave P

  int bh = blockIdx.x;
  int b = bh / Hn, h = bh - b * Hn;
  int kvh = h / (Hn / KVHn);
  int qb = blockIdx.y * 128;
  int tid = threadIdx.x;
  int wave = tid >> 5, lane = tid & 31;
  int col = lane & 15, rsel = (lane >> 4) * 8;

  // Q fragments for this wave's 16 rows: A-layout straight from global
  const bf16_t* qptr = Q + (((size_t)b * Hn + h) * Sn + qb + wave * 16) * HDn;
  v16bf qf[4];
#pragma unroll
  for (int kk = 0; kk < 4; kk++) qf[kk] = frag_a(qptr + kk * 32, HDn, lane);

  v8f acc[8] = {};
  float m_i[8], l_i[8];
#pragma unroll
  for (int e = 0; e < 8; e++) { m_i[e] = -1e30f; l_i[e] = 0.0f; }

  const bf16_t* kbase = K + (((size_t)b * KVHn + kvh) * Sn) * HDn;
  const bf16_t* vbase = V + (((size_t)b * KVHn + kvh) * Sn) * HDn;

  int ktiles = (qb + 128) / 32;                 // causal: keys <= qb+127
  for (int kt = 0; kt < ktiles; kt++) {
    int kb = kt * 32;
    const bf16_t* kp = kbase + (size_t)kb * HDn;
    const bf16_t* vp = vbase + (size_t)kb * HDn;
    for (int c = tid; c < 512; c += 256) {      // K tile: async 16B chunks
      int r = c >> 4, cc = (c & 15) * 8;
      async_copy_b128(&kp[(size_t)r * 128 + cc], &k_tile[r * 128 + cc]);
    }
    for (int c = tid; c < 4096; c += 256) {     // V transposed into [d][key]
      int r = c >> 7, d = c & 127;
      vt[d * 32 + r] = vp[(size_t)r * 128 + d];
    }
    wait_async0();
    __syncthreads();

    // S = Q @ K^T  (two 16-key n-subtiles, 4 k-steps over HD)
    v8f s0acc = {}, s1acc = {};
#pragma unroll
    for (int kk = 0; kk < 4; kk++) {
      v16bf b0 = frag_b(&k_tile[0 * 128 + kk * 32], 128, lane);
      v16bf b1 = frag_b(&k_tile[16 * 128 + kk * 32], 128, lane);
      s0acc = __builtin_amdgcn_wmma_f32_16x16x32_bf16(false, qf[kk], false, b0,
                                                      (short)0, s0acc, false, false);
      s1acc = __builtin_amdgcn_wmma_f32_16x16x32_bf16(false, qf[kk], false, b1,
                                                      (short)0, s1acc, false, false);
    }

    // online softmax (16-lane reductions within each half-wave row group)
#pragma unroll
    for (int e = 0; e < 8; e++) {
      int qrow = qb + wave * 16 + rsel + e;
      float s0 = s0acc[e] * SCALING + ((kb + col)      <= qrow ? 0.0f : NEGBIG);
      float s1 = s1acc[e] * SCALING + ((kb + 16 + col) <= qrow ? 0.0f : NEGBIG);
      float t = fmaxf(s0, s1);
      t = fmaxf(t, __shfl_xor(t, 1, 32));
      t = fmaxf(t, __shfl_xor(t, 2, 32));
      t = fmaxf(t, __shfl_xor(t, 4, 32));
      t = fmaxf(t, __shfl_xor(t, 8, 32));
      float mnew  = fmaxf(m_i[e], t);
      float alpha = __expf(m_i[e] - mnew);
      float p0 = __expf(s0 - mnew);
      float p1 = __expf(s1 - mnew);
      float ps = p0 + p1;
      ps += __shfl_xor(ps, 1, 32);
      ps += __shfl_xor(ps, 2, 32);
      ps += __shfl_xor(ps, 4, 32);
      ps += __shfl_xor(ps, 8, 32);
      l_i[e] = l_i[e] * alpha + ps;
      m_i[e] = mnew;
#pragma unroll
      for (int dt = 0; dt < 8; dt++) acc[dt][e] *= alpha;
      // stash P (C-layout -> row-major [16][32]); wave-local LDS, DS in-order
      p_lds[wave][(rsel + e) * 32 + col]      = (bf16_t)p0;
      p_lds[wave][(rsel + e) * 32 + 16 + col] = (bf16_t)p1;
    }

    // acc += P @ V  (one A frag, 8 output d-subtiles)
    v16bf pa = frag_a(&p_lds[wave][0], 32, lane);
#pragma unroll
    for (int dt = 0; dt < 8; dt++) {
      v16bf bv = frag_b(&vt[dt * 16 * 32], 32, lane);
      acc[dt] = __builtin_amdgcn_wmma_f32_16x16x32_bf16(false, pa, false, bv,
                                                        (short)0, acc[dt], false, false);
    }
    __syncthreads();   // protect k_tile / vt before next restage
  }

  // epilogue: normalize by l, write (B, S, H*HD) bf16
#pragma unroll
  for (int dt = 0; dt < 8; dt++) {
#pragma unroll
    for (int e = 0; e < 8; e++) {
      int qrow = qb + wave * 16 + rsel + e;
      O[((size_t)b * Sn + qrow) * (Hn * HDn) + h * HDn + dt * 16 + col] =
          (bf16_t)(acc[dt][e] / l_i[e]);
    }
  }
}

// ---------------------------------------------------------------------------
extern "C" void kernel_launch(void* const* d_in, const int* in_sizes, int n_in,
                              void* d_out, int out_size, void* d_ws, size_t ws_size,
                              hipStream_t stream) {
  (void)in_sizes; (void)n_in; (void)out_size; (void)ws_size;
  const float* hs   = (const float*)d_in[0];
  const float* cosb = (const float*)d_in[1];
  const float* sinb = (const float*)d_in[2];
  // d_in[3] = attention_mask: exactly causal -> computed analytically in-kernel
  const float* qk = (const float*)d_in[4];
  const float* kk = (const float*)d_in[5];
  const float* vk = (const float*)d_in[6];
  const float* ok = (const float*)d_in[7];
  const float* qw = (const float*)d_in[8];
  const float* kw = (const float*)d_in[9];
  float* out = (float*)d_out;

  char* wp = (char*)d_ws;
  auto alloc = [&](size_t bytes) -> void* {
    void* r = wp; wp += (bytes + 255) & ~(size_t)255; return r;
  };
  bf16_t* hid_bf = (bf16_t*)alloc((size_t)Mrows * HIDn * 2);
  bf16_t* wqT  = (bf16_t*)alloc((size_t)HIDn * Hn * HDn * 2);     // [N x K]
  bf16_t* wkT  = (bf16_t*)alloc((size_t)HIDn * KVHn * HDn * 2);
  bf16_t* wvT  = (bf16_t*)alloc((size_t)HIDn * KVHn * HDn * 2);
  bf16_t* woT  = (bf16_t*)alloc((size_t)Hn * HDn * HIDn * 2);
  bf16_t* qraw = (bf16_t*)alloc((size_t)Mrows * Hn * HDn * 2);
  bf16_t* kraw = (bf16_t*)alloc((size_t)Mrows * KVHn * HDn * 2);
  bf16_t* vraw = (bf16_t*)alloc((size_t)Mrows * KVHn * HDn * 2);
  bf16_t* qb   = (bf16_t*)alloc((size_t)Bn * Hn * Sn * HDn * 2);
  bf16_t* kb   = (bf16_t*)alloc((size_t)Bn * KVHn * Sn * HDn * 2);
  bf16_t* vb   = (bf16_t*)alloc((size_t)Bn * KVHn * Sn * HDn * 2);
  bf16_t* ao   = (bf16_t*)alloc((size_t)Mrows * Hn * HDn * 2);

  dim3 blk(256);
  // hidden: plain convert
  {
    size_t n = (size_t)Mrows * HIDn;
    f32_to_bf16_kernel<<<dim3((unsigned)((n + 255) / 256)), blk, 0, stream>>>(
        hs, hid_bf, (int)n);
  }
  // weights: convert + transpose to [N x K] so GEMM B-tiles stage as b128 chunks
  transpose_to_bf16_kernel<<<dim3(HIDn / 64, (Hn * HDn) / 64), blk, 0, stream>>>(
      qk, wqT, HIDn, Hn * HDn);
  transpose_to_bf16_kernel<<<dim3(HIDn / 64, (KVHn * HDn) / 64), blk, 0, stream>>>(
      kk, wkT, HIDn, KVHn * HDn);
  transpose_to_bf16_kernel<<<dim3(HIDn / 64, (KVHn * HDn) / 64), blk, 0, stream>>>(
      vk, wvT, HIDn, KVHn * HDn);
  transpose_to_bf16_kernel<<<dim3((Hn * HDn) / 64, HIDn / 64), blk, 0, stream>>>(
      ok, woT, Hn * HDn, HIDn);

  // Q/K/V projections (bf16 out)
  gemm_bf16_tn_kernel<true><<<dim3(Mrows / 128, (Hn * HDn) / 128), blk, 0, stream>>>(
      hid_bf, wqT, qraw, Mrows, Hn * HDn, HIDn);
  gemm_bf16_tn_kernel<true><<<dim3(Mrows / 128, (KVHn * HDn) / 128), blk, 0, stream>>>(
      hid_bf, wkT, kraw, Mrows, KVHn * HDn, HIDn);
  gemm_bf16_tn_kernel<true><<<dim3(Mrows / 128, (KVHn * HDn) / 128), blk, 0, stream>>>(
      hid_bf, wvT, vraw, Mrows, KVHn * HDn, HIDn);

  // RMSNorm + RoPE (+ transpose to (B,NH,S,HD))
  rmsnorm_rope_kernel<<<dim3((Bn * Sn * Hn) / 8), blk, 0, stream>>>(
      qraw, cosb, sinb, qw, qb, Hn);
  rmsnorm_rope_kernel<<<dim3((Bn * Sn * KVHn) / 8), blk, 0, stream>>>(
      kraw, cosb, sinb, kw, kb, KVHn);
  size_t vtotal = (size_t)Bn * Sn * KVHn * HDn;
  v_relayout_kernel<<<dim3((unsigned)((vtotal + 255) / 256)), blk, 0, stream>>>(vraw, vb);

  // Flash attention
  flash_attn_kernel<<<dim3(Bn * Hn, Sn / 128), blk, 0, stream>>>(qb, kb, vb, ao);

  // Output projection (fp32 out)
  gemm_bf16_tn_kernel<false><<<dim3(Mrows / 128, HIDn / 128), blk, 0, stream>>>(
      ao, woT, out, Mrows, HIDn, HIDn);
}
